// StandardDecoderBlock_58274116272260
// MI455X (gfx1250) — matrix operand order repaired and checked by reading, not compiled, not run on hardware
//
#include <hip/hip_runtime.h>
#include <hip/hip_bf16.h>

typedef unsigned short u16;
typedef __attribute__((ext_vector_type(8)))  int    v8i;
typedef __attribute__((ext_vector_type(8)))  float  v8f;
typedef __attribute__((ext_vector_type(16))) __bf16 v16bf;

#define DEV __device__ __forceinline__

// ---------------- problem dims ----------------
#define DB   2
#define DS   1024
#define DT   1024
#define DD   768
#define DH_  12
#define DG   4
#define DDH  64
#define DF   3072
#define DE   8
#define NTOK (DB * DS)     // 2048 (both x and encoder have 2048 tokens)
#define QLD  (DH_ * DDH)   // 768
#define KVLD (DG * DDH)    // 256

// ---------------- helpers ----------------
DEV u16 f2bf(float f) {
  unsigned u = __float_as_uint(f);
  unsigned r = u + 0x7FFFu + ((u >> 16) & 1u);   // round-to-nearest-even
  return (u16)(r >> 16);
}
DEV float bf2f(u16 h) { return __uint_as_float(((unsigned)h) << 16); }

DEV float gelu_tanh(float x) {
  float x3 = x * x * x;
  return 0.5f * x * (1.f + tanhf(0.79788456f * (x + 0.044715f * x3)));
}

DEV v8f wmma_bf16(v8i a, v8i b, v8f c) {
  return __builtin_amdgcn_wmma_f32_16x16x32_bf16(
      false, __builtin_bit_cast(v16bf, a),
      false, __builtin_bit_cast(v16bf, b),
      (short)0, c, false, false);
}

// A fragment (16x32 bf16). lane holds row m = lane&15.
// k(v,s) = kbase + 16*half + 2*v + s  (pairs contiguous in memory)
DEV v8i load_a_frag(const u16* rowp, int kbase, int half) {
  const u16* p = rowp + kbase + half * 16;
  v8i f;
  f[0] = *(const int*)(p + 0);  f[1] = *(const int*)(p + 2);
  f[2] = *(const int*)(p + 4);  f[3] = *(const int*)(p + 6);
  f[4] = *(const int*)(p + 8);  f[5] = *(const int*)(p + 10);
  f[6] = *(const int*)(p + 12); f[7] = *(const int*)(p + 14);
  return f;
}

// B fragment (32x16 bf16) from K-major storage (Bt[n][k], row = this lane's column n).
// k(v,s) = kbase + 16*half + 2*v + s
DEV v8i load_b_frag(const u16* rowp, int kbase, int half) {
  const u16* p = rowp + kbase + half * 16;
  v8i f;
#pragma unroll
  for (int v = 0; v < 8; ++v) f[v] = *(const int*)(p + 2 * v);
  return f;
}

// B fragment gathered from row-major [k][n] storage (rows strided by ld ushorts).
DEV v8i load_b_gather(const u16* colp, int ld, int half) {
  const u16* p = colp + (size_t)(half * 16) * ld;
  v8i f;
#pragma unroll
  for (int v = 0; v < 8; ++v) {
    unsigned lo = p[(size_t)(2 * v) * ld];
    unsigned hi = p[(size_t)(2 * v + 1) * ld];
    f[v] = (int)(lo | (hi << 16));
  }
  return f;
}

// ---------------- conversion kernels ----------------
__global__ void cvt_transpose(const float* __restrict__ W, u16* __restrict__ Wt,
                              int K, int N) {
  long long idx = (long long)blockIdx.x * blockDim.x + threadIdx.x;
  if (idx >= (long long)K * N) return;
  int k = (int)(idx / N), n = (int)(idx % N);
  Wt[(size_t)n * K + k] = f2bf(W[idx]);
}

__global__ void cvt_bf16(const float* __restrict__ A, u16* __restrict__ B, long long n) {
  long long i = (long long)blockIdx.x * blockDim.x + threadIdx.x;
  if (i < n) B[i] = f2bf(A[i]);
}

// ---------------- LayerNorm (block per token, D=768) ----------------
__global__ void ln_bf16(const float* __restrict__ x, const float* __restrict__ sc,
                        const float* __restrict__ bi, u16* __restrict__ out) {
  __shared__ float red[256];
  int t = blockIdx.x, tid = threadIdx.x;
  const float* xr = x + (size_t)t * DD;
  float v0 = xr[tid], v1 = xr[tid + 256], v2 = xr[tid + 512];
  red[tid] = v0 + v1 + v2;
  __syncthreads();
  for (int o = 128; o > 0; o >>= 1) { if (tid < o) red[tid] += red[tid + o]; __syncthreads(); }
  float mu = red[0] / (float)DD;
  __syncthreads();
  float d0 = v0 - mu, d1 = v1 - mu, d2 = v2 - mu;
  red[tid] = d0 * d0 + d1 * d1 + d2 * d2;
  __syncthreads();
  for (int o = 128; o > 0; o >>= 1) { if (tid < o) red[tid] += red[tid + o]; __syncthreads(); }
  float rs = rsqrtf(red[0] / (float)DD + 1e-6f);
  u16* orow = out + (size_t)t * DD;
  orow[tid]       = f2bf(d0 * rs * sc[tid]       + bi[tid]);
  orow[tid + 256] = f2bf(d1 * rs * sc[tid + 256] + bi[tid + 256]);
  orow[tid + 512] = f2bf(d2 * rs * sc[tid + 512] + bi[tid + 512]);
}

// ---------------- generic WMMA GEMM: out = A[M,K] * Bt[N,K]^T + bias ----------------
// Block = 256 threads = 8 waves stacked in M. Block tile 128(m) x 64(n).
// Each wave owns a 16x64 strip: one A fragment is reused across 4 B tiles
// (4 accumulators), quadrupling WMMA work per byte of A traffic.
// mode 0: outb = bf16(acc+bias)
// mode 1: outf = res + acc + bias                     (residual add, f32)
// mode 2: outb = bf16(gelu(acc+bias))
// mode 3: outf += rowscale[m*rs_stride]*(acc+bias)    (MoE combine accumulate)
__global__ void gemm_wmma(const u16* __restrict__ A, const u16* __restrict__ Bt,
                          const float* __restrict__ bias,
                          u16* __restrict__ outb, float* __restrict__ outf,
                          const float* __restrict__ res,
                          const float* __restrict__ rowscale, int rs_stride,
                          int M, int N, int Kd, int mode) {
  int tid = threadIdx.x, lane = tid & 31, w = tid >> 5;
  int m0 = blockIdx.y * 128 + w * 16;
  int n0 = blockIdx.x * 64;
  if (m0 >= M || n0 >= N) return;          // wave-uniform
  int half = (lane >> 4) & 1, nl = lane & 15;
  const u16* arow = A + (size_t)(m0 + nl) * Kd;
  const u16* brow[4];
#pragma unroll
  for (int t = 0; t < 4; ++t) brow[t] = Bt + (size_t)(n0 + t * 16 + nl) * Kd;
  v8f acc[4];
#pragma unroll
  for (int t = 0; t < 4; ++t) acc[t] = (v8f){0.f,0.f,0.f,0.f,0.f,0.f,0.f,0.f};
  for (int k = 0; k < Kd; k += 32) {
    v8i a = load_a_frag(arow, k, half);
#pragma unroll
    for (int t = 0; t < 4; ++t)
      acc[t] = wmma_bf16(a, load_b_frag(brow[t], k, half), acc[t]);
  }
#pragma unroll
  for (int t = 0; t < 4; ++t) {
    int ncol = n0 + t * 16 + nl;
    float bn = bias[ncol];
#pragma unroll
    for (int r = 0; r < 8; ++r) {
      int m = m0 + r + 8 * half;
      size_t idx = (size_t)m * N + ncol;
      float val = acc[t][r] + bn;
      if (mode == 0)      outb[idx] = f2bf(val);
      else if (mode == 1) outf[idx] = res[idx] + val;
      else if (mode == 2) outb[idx] = f2bf(gelu_tanh(val));
      else                outf[idx] += rowscale[(size_t)m * rs_stride] * val;
    }
  }
}

// ---------------- flash attention: 1 wave per (b, h, 16-query tile) ----------------
__global__ void flash_attn(const u16* __restrict__ Q, const u16* __restrict__ Kb,
                           const u16* __restrict__ Vb, u16* __restrict__ O,
                           int Tkv, int causal) {
  __shared__ u16 pshm[16 * 36];           // 16x32 P tile, padded row stride
  int wid = blockIdx.x;
  int qt = wid & ((DS / 16) - 1); wid >>= 6;     // DS/16 = 64
  int h  = wid % DH_;
  int b  = wid / DH_;
  int g  = h / (DH_ / DG);
  int lane = threadIdx.x;
  int half = (lane >> 4) & 1, nl = lane & 15;

  const u16* qrow = Q + (size_t)(b * DS + qt * 16 + nl) * QLD + h * DDH;
  v8i qa0 = load_a_frag(qrow, 0, half);
  v8i qa1 = load_a_frag(qrow, 32, half);

  float mrow[8], lrow[8];
  v8f o0 = {0.f,0.f,0.f,0.f,0.f,0.f,0.f,0.f}, o1 = o0, o2 = o0, o3 = o0;
#pragma unroll
  for (int r = 0; r < 8; ++r) { mrow[r] = -1e30f; lrow[r] = 0.f; }

  int nblk = causal ? (qt * 16 + 47) / 32 : (Tkv + 31) / 32;
  for (int blk = 0; blk < nblk; ++blk) {
    int tb = blk * 32;
    // scores = Q (16x64) * K^T : K^T is Bt[n=t][k=d]; K buffer is t-major, d contiguous
    const u16* r0 = Kb + (size_t)(b * DT + tb + nl) * KVLD + g * DDH;
    const u16* r1 = r0 + (size_t)16 * KVLD;
    v8f s0 = {0.f,0.f,0.f,0.f,0.f,0.f,0.f,0.f}, s1 = s0;
    s0 = wmma_bf16(qa0, load_b_frag(r0, 0, half), s0);
    s0 = wmma_bf16(qa1, load_b_frag(r0, 32, half), s0);
    s1 = wmma_bf16(qa0, load_b_frag(r1, 0, half), s1);
    s1 = wmma_bf16(qa1, load_b_frag(r1, 32, half), s1);
#pragma unroll
    for (int r = 0; r < 8; ++r) {
      int qg = qt * 16 + r + 8 * half;
      float a0 = s0[r] * 0.125f, a1 = s1[r] * 0.125f;
      if (causal) {
        if (tb + nl      > qg) a0 = -1e30f;
        if (tb + 16 + nl > qg) a1 = -1e30f;
      }
      float lm = fmaxf(a0, a1);
      lm = fmaxf(lm, __shfl_xor(lm, 1, 32));
      lm = fmaxf(lm, __shfl_xor(lm, 2, 32));
      lm = fmaxf(lm, __shfl_xor(lm, 4, 32));
      lm = fmaxf(lm, __shfl_xor(lm, 8, 32));
      float nm = fmaxf(mrow[r], lm);
      float p0 = __expf(a0 - nm), p1 = __expf(a1 - nm);
      float ls = p0 + p1;
      ls += __shfl_xor(ls, 1, 32);
      ls += __shfl_xor(ls, 2, 32);
      ls += __shfl_xor(ls, 4, 32);
      ls += __shfl_xor(ls, 8, 32);
      float al = __expf(mrow[r] - nm);
      lrow[r] = lrow[r] * al + ls;
      mrow[r] = nm;
      o0[r] *= al; o1[r] *= al; o2[r] *= al; o3[r] *= al;
      int row = r + 8 * half;
      pshm[row * 36 + nl]      = f2bf(p0);
      pshm[row * 36 + 16 + nl] = f2bf(p1);
    }
    __syncthreads();
    // P fragment (A layout, 16x32) from LDS
    v8i pa;
    {
      const u16* p = &pshm[(lane & 15) * 36 + half * 16];
      pa[0] = *(const int*)(p + 0);  pa[1] = *(const int*)(p + 2);
      pa[2] = *(const int*)(p + 4);  pa[3] = *(const int*)(p + 6);
      pa[4] = *(const int*)(p + 8);  pa[5] = *(const int*)(p + 10);
      pa[6] = *(const int*)(p + 12); pa[7] = *(const int*)(p + 14);
    }
    const u16* vbase = Vb + (size_t)(b * DT + tb) * KVLD + g * DDH + nl;
    o0 = wmma_bf16(pa, load_b_gather(vbase + 0,  KVLD, half), o0);
    o1 = wmma_bf16(pa, load_b_gather(vbase + 16, KVLD, half), o1);
    o2 = wmma_bf16(pa, load_b_gather(vbase + 32, KVLD, half), o2);
    o3 = wmma_bf16(pa, load_b_gather(vbase + 48, KVLD, half), o3);
    __syncthreads();
  }
#pragma unroll
  for (int r = 0; r < 8; ++r) {
    float inv = 1.f / lrow[r];
    int row = r + 8 * half;
    u16* orow = O + (size_t)(b * DS + qt * 16 + row) * QLD + h * DDH;
    orow[nl]      = f2bf(o0[r] * inv);
    orow[16 + nl] = f2bf(o1[r] * inv);
    orow[32 + nl] = f2bf(o2[r] * inv);
    orow[48 + nl] = f2bf(o3[r] * inv);
  }
}

// ---------------- MoE gating ----------------
__global__ void gate_logits(const u16* __restrict__ xn, const float* __restrict__ gw,
                            const float* __restrict__ gb, float* __restrict__ logits) {
  int idx = blockIdx.x * blockDim.x + threadIdx.x;
  if (idx >= NTOK * DE) return;
  int t = idx >> 3, e = idx & 7;
  float s = gb[e];
  const u16* xr = xn + (size_t)t * DD;
  for (int d = 0; d < DD; ++d) s += bf2f(xr[d]) * gw[d * DE + e];
  logits[idx] = s;
}

__global__ void gate_combine(const float* __restrict__ logits, float* __restrict__ comb) {
  int t = blockIdx.x * blockDim.x + threadIdx.x;
  if (t >= NTOK) return;
  float l[DE]; float mx = -1e30f;
  for (int e = 0; e < DE; ++e) { l[e] = logits[t * DE + e]; mx = fmaxf(mx, l[e]); }
  float s = 0.f;
  for (int e = 0; e < DE; ++e) { l[e] = __expf(l[e] - mx); s += l[e]; }
  for (int e = 0; e < DE; ++e) l[e] /= s;
  int i1 = 0;
  for (int e = 1; e < DE; ++e) if (l[e] > l[i1]) i1 = e;
  int i2 = (i1 == 0) ? 1 : 0;
  for (int e = 0; e < DE; ++e) if (e != i1 && l[e] > l[i2]) i2 = e;
  float denom = l[i1] + l[i2];
  for (int e = 0; e < DE; ++e)
    comb[t * DE + e] = (e == i1) ? l[i1] / denom : ((e == i2) ? l[i2] / denom : 0.f);
}

// ---------------- host launch ----------------
extern "C" void kernel_launch(void* const* d_in, const int* in_sizes, int n_in,
                              void* d_out, int out_size, void* d_ws, size_t ws_size,
                              hipStream_t stream) {
  (void)in_sizes; (void)n_in; (void)out_size; (void)ws_size;
  const float* x     = (const float*)d_in[0];
  const float* enc   = (const float*)d_in[1];
  const float* ln1s  = (const float*)d_in[2];
  const float* ln1b  = (const float*)d_in[3];
  const float* ln2s  = (const float*)d_in[4];
  const float* ln2b  = (const float*)d_in[5];
  const float* ln3s  = (const float*)d_in[6];
  const float* ln3b  = (const float*)d_in[7];
  const float* sa_wq = (const float*)d_in[8];  const float* sa_bq = (const float*)d_in[9];
  const float* sa_wk = (const float*)d_in[10]; const float* sa_bk = (const float*)d_in[11];
  const float* sa_wv = (const float*)d_in[12]; const float* sa_bv = (const float*)d_in[13];
  const float* sa_wo = (const float*)d_in[14]; const float* sa_bo = (const float*)d_in[15];
  const float* ca_wq = (const float*)d_in[16]; const float* ca_bq = (const float*)d_in[17];
  const float* ca_wk = (const float*)d_in[18]; const float* ca_bk = (const float*)d_in[19];
  const float* ca_wv = (const float*)d_in[20]; const float* ca_bv = (const float*)d_in[21];
  const float* ca_wo = (const float*)d_in[22]; const float* ca_bo = (const float*)d_in[23];
  const float* gw    = (const float*)d_in[24]; const float* gb    = (const float*)d_in[25];
  const float* w1    = (const float*)d_in[26]; const float* b1    = (const float*)d_in[27];
  const float* w2    = (const float*)d_in[28]; const float* b2    = (const float*)d_in[29];
  float* out = (float*)d_out;

  char* ws = (char*)d_ws;
  size_t off = 0;
  auto take = [&](size_t bytes) -> void* {
    off = (off + 255) & ~(size_t)255;
    void* p = ws + off; off += bytes; return p;
  };
  u16* xnb   = (u16*)take((size_t)NTOK * DD * 2);
  u16* encb  = (u16*)take((size_t)NTOK * DD * 2);
  u16* qb    = (u16*)take((size_t)NTOK * QLD * 2);
  u16* kb    = (u16*)take((size_t)NTOK * KVLD * 2);
  u16* vb    = (u16*)take((size_t)NTOK * KVLD * 2);
  u16* aob   = (u16*)take((size_t)NTOK * QLD * 2);
  u16* hb    = (u16*)take((size_t)NTOK * DF * 2);
  float* lg  = (float*)take((size_t)NTOK * DE * 4);
  float* cmb = (float*)take((size_t)NTOK * DE * 4);
  u16* saqT  = (u16*)take((size_t)DD * QLD * 2);
  u16* sakT  = (u16*)take((size_t)DD * KVLD * 2);
  u16* savT  = (u16*)take((size_t)DD * KVLD * 2);
  u16* saoT  = (u16*)take((size_t)QLD * DD * 2);
  u16* caqT  = (u16*)take((size_t)DD * QLD * 2);
  u16* cakT  = (u16*)take((size_t)DD * KVLD * 2);
  u16* cavT  = (u16*)take((size_t)DD * KVLD * 2);
  u16* caoT  = (u16*)take((size_t)QLD * DD * 2);
  u16* w1T   = (u16*)take((size_t)DE * DD * DF * 2);
  u16* w2T   = (u16*)take((size_t)DE * DF * DD * 2);

  auto cgrid = [](long long n) { return dim3((unsigned)((n + 255) / 256)); };
  auto ggrid = [](int M, int N) { return dim3((unsigned)((N + 63) / 64), (unsigned)((M + 127) / 128)); };

  // weight + context conversions
  cvt_transpose<<<cgrid((long long)DD * QLD), 256, 0, stream>>>(sa_wq, saqT, DD, QLD);
  cvt_transpose<<<cgrid((long long)DD * KVLD), 256, 0, stream>>>(sa_wk, sakT, DD, KVLD);
  cvt_transpose<<<cgrid((long long)DD * KVLD), 256, 0, stream>>>(sa_wv, savT, DD, KVLD);
  cvt_transpose<<<cgrid((long long)QLD * DD), 256, 0, stream>>>(sa_wo, saoT, QLD, DD);
  cvt_transpose<<<cgrid((long long)DD * QLD), 256, 0, stream>>>(ca_wq, caqT, DD, QLD);
  cvt_transpose<<<cgrid((long long)DD * KVLD), 256, 0, stream>>>(ca_wk, cakT, DD, KVLD);
  cvt_transpose<<<cgrid((long long)DD * KVLD), 256, 0, stream>>>(ca_wv, cavT, DD, KVLD);
  cvt_transpose<<<cgrid((long long)QLD * DD), 256, 0, stream>>>(ca_wo, caoT, QLD, DD);
  for (int e = 0; e < DE; ++e) {
    cvt_transpose<<<cgrid((long long)DD * DF), 256, 0, stream>>>(
        w1 + (size_t)e * DD * DF, w1T + (size_t)e * DF * DD, DD, DF);
    cvt_transpose<<<cgrid((long long)DF * DD), 256, 0, stream>>>(
        w2 + (size_t)e * DF * DD, w2T + (size_t)e * DD * DF, DF, DD);
  }
  cvt_bf16<<<cgrid((long long)NTOK * DD), 256, 0, stream>>>(enc, encb, (long long)NTOK * DD);

  dim3 blk(256);
  unsigned attn_grid = DB * DH_ * (DS / 16);

  // ---- self attention ----
  ln_bf16<<<NTOK, 256, 0, stream>>>(x, ln1s, ln1b, xnb);
  gemm_wmma<<<ggrid(NTOK, QLD), blk, 0, stream>>>(xnb, saqT, sa_bq, qb, nullptr, nullptr, nullptr, 0, NTOK, QLD, DD, 0);
  gemm_wmma<<<ggrid(NTOK, KVLD), blk, 0, stream>>>(xnb, sakT, sa_bk, kb, nullptr, nullptr, nullptr, 0, NTOK, KVLD, DD, 0);
  gemm_wmma<<<ggrid(NTOK, KVLD), blk, 0, stream>>>(xnb, savT, sa_bv, vb, nullptr, nullptr, nullptr, 0, NTOK, KVLD, DD, 0);
  flash_attn<<<attn_grid, 32, 0, stream>>>(qb, kb, vb, aob, DS, 1);
  gemm_wmma<<<ggrid(NTOK, DD), blk, 0, stream>>>(aob, saoT, sa_bo, nullptr, out, x, nullptr, 0, NTOK, DD, QLD, 1);

  // ---- cross attention ----
  ln_bf16<<<NTOK, 256, 0, stream>>>(out, ln2s, ln2b, xnb);
  gemm_wmma<<<ggrid(NTOK, QLD), blk, 0, stream>>>(xnb, caqT, ca_bq, qb, nullptr, nullptr, nullptr, 0, NTOK, QLD, DD, 0);
  gemm_wmma<<<ggrid(NTOK, KVLD), blk, 0, stream>>>(encb, cakT, ca_bk, kb, nullptr, nullptr, nullptr, 0, NTOK, KVLD, DD, 0);
  gemm_wmma<<<ggrid(NTOK, KVLD), blk, 0, stream>>>(encb, cavT, ca_bv, vb, nullptr, nullptr, nullptr, 0, NTOK, KVLD, DD, 0);
  flash_attn<<<attn_grid, 32, 0, stream>>>(qb, kb, vb, aob, DT, 0);
  gemm_wmma<<<ggrid(NTOK, DD), blk, 0, stream>>>(aob, caoT, ca_bo, nullptr, out, out, nullptr, 0, NTOK, DD, QLD, 1);

  // ---- MoE ----
  ln_bf16<<<NTOK, 256, 0, stream>>>(out, ln3s, ln3b, xnb);
  gate_logits<<<cgrid((long long)NTOK * DE), 256, 0, stream>>>(xnb, gw, gb, lg);
  gate_combine<<<cgrid(NTOK), 256, 0, stream>>>(lg, cmb);
  for (int e = 0; e < DE; ++e) {
    gemm_wmma<<<ggrid(NTOK, DF), blk, 0, stream>>>(
        xnb, w1T + (size_t)e * DF * DD, b1 + (size_t)e * DF,
        hb, nullptr, nullptr, nullptr, 0, NTOK, DF, DD, 2);
    gemm_wmma<<<ggrid(NTOK, DD), blk, 0, stream>>>(
        hb, w2T + (size_t)e * DD * DF, b2 + (size_t)e * DD,
        nullptr, out, nullptr, cmb + e, DE, NTOK, DD, DF, 3);
  }
}
